// GRU_84490596647263
// MI455X (gfx1250) — compile-verified, matrix operand order checked by reading, not run on hardware
//
#include <hip/hip_runtime.h>
#include <cstdint>

// ---------------------------------------------------------------------------
// 2-layer GRU (B=64, T=512, D=H=1024) + FC(1) for MI455X (gfx1250).
// bf16 WMMA (v_wmma_f32_16x16x32_bf16) with fp32 accumulation; weights
// converted once to bf16 (~25MB, L2-resident in 192MB L2). Sequential T
// recurrence = 2 stream-ordered kernels per step (graph-replayed).
// Each wave32 owns one 16x16 tile of h_new with 4 accumulators
// (r-sum, z-sum, i_n, h_n); the K loops are software-pipelined with
// register double-buffering so loads for chunk k+32 are in flight while
// WMMAs consume chunk k (partial s_wait_loadcnt instead of full drains).
// ---------------------------------------------------------------------------

#define Bdim   64
#define Tdim   512
#define Ddim   1024
#define Hdim   1024

typedef unsigned short u16;
typedef __attribute__((ext_vector_type(16))) __bf16 v16bf;
typedef __attribute__((ext_vector_type(8)))  float  v8f;

#define WMMA_BF16(a, b, c) \
    __builtin_amdgcn_wmma_f32_16x16x32_bf16(false, (a), false, (b), (short)0, (c), false, false)

__device__ __forceinline__ float bf2f(u16 h) {
    return __uint_as_float(((unsigned int)h) << 16);
}
__device__ __forceinline__ u16 f2bf(float f) {
    unsigned int u = __float_as_uint(f);
    unsigned int r = u + 0x7fffu + ((u >> 16) & 1u);   // round-to-nearest-even
    return (u16)(r >> 16);
}
__device__ __forceinline__ float sigmoidf_fast(float x) {
    return 1.0f / (1.0f + __expf(-x));
}

// Per-lane fragment of a 16x32 bf16 WMMA tile: 16 contiguous bf16 = 2 x b128.
// Lane mapping (contiguous k order used here, A and B alike): lane%16 = tile row (A) / output
// column (B); lane/16 = K-half. Caller bakes both into the pointer/offset.
__device__ __forceinline__ v16bf load16(const u16* __restrict__ p) {
    const uint4* q = (const uint4*)p;
    union { uint4 u[2]; v16bf v; } t;
    t.u[0] = q[0];
    t.u[1] = q[1];
    return t.v;
}

// One GEMM phase over reduction length K (multiple of 64): accumulates
//   rs  += A . Wr^T,  zs += A . Wz^T,  nacc += A . Wn^T
// with ping-pong register buffers so each WMMA triple overlaps the next
// chunk's 8 b128 loads.
__device__ __forceinline__ void gemm_phase(const u16* __restrict__ arow,
                                           const u16* __restrict__ wr,
                                           const u16* __restrict__ wz,
                                           const u16* __restrict__ wn,
                                           int K, int kofs,
                                           v8f& rs, v8f& zs, v8f& nacc) {
    v16bf a0  = load16(arow + kofs);
    v16bf br0 = load16(wr   + kofs);
    v16bf bz0 = load16(wz   + kofs);
    v16bf bn0 = load16(wn   + kofs);
    v16bf a1, br1, bz1, bn1;

    int k = 0;
    for (; k < K - 64; k += 64) {
        const int k1 = k + 32 + kofs;
        a1  = load16(arow + k1);
        br1 = load16(wr   + k1);
        bz1 = load16(wz   + k1);
        bn1 = load16(wn   + k1);
        rs   = WMMA_BF16(a0, br0, rs);
        zs   = WMMA_BF16(a0, bz0, zs);
        nacc = WMMA_BF16(a0, bn0, nacc);

        const int k2 = k + 64 + kofs;
        a0  = load16(arow + k2);
        br0 = load16(wr   + k2);
        bz0 = load16(wz   + k2);
        bn0 = load16(wn   + k2);
        rs   = WMMA_BF16(a1, br1, rs);
        zs   = WMMA_BF16(a1, bz1, zs);
        nacc = WMMA_BF16(a1, bn1, nacc);
    }
    // Last two chunks (k and k+32)
    const int k1 = k + 32 + kofs;
    a1  = load16(arow + k1);
    br1 = load16(wr   + k1);
    bz1 = load16(wz   + k1);
    bn1 = load16(wn   + k1);
    rs   = WMMA_BF16(a0, br0, rs);
    zs   = WMMA_BF16(a0, bz0, zs);
    nacc = WMMA_BF16(a0, bn0, nacc);
    rs   = WMMA_BF16(a1, br1, rs);
    zs   = WMMA_BF16(a1, bz1, zs);
    nacc = WMMA_BF16(a1, bn1, nacc);
}

// ---------------------------------------------------------------------------
// Elementwise helpers
// ---------------------------------------------------------------------------
__global__ void f32_to_bf16_kernel(const float* __restrict__ src,
                                   u16* __restrict__ dst, int n) {
    int i = blockIdx.x * blockDim.x + threadIdx.x;
    if (i < n) dst[i] = f2bf(src[i]);
}

__global__ void zero_u16_kernel(u16* __restrict__ p, int n) {
    int i = blockIdx.x * blockDim.x + threadIdx.x;
    if (i < n) p[i] = 0;
}

// ---------------------------------------------------------------------------
// One GRU layer-step:  h_new = GRUCell(x_t, h_prev)
// Grid: 32 blocks x 256 threads = 256 waves; wave -> 16x16 h_new tile.
// ---------------------------------------------------------------------------
__global__ __launch_bounds__(256)
void gru_step_kernel(const u16* __restrict__ xsrc, size_t xRowStride, int Din,
                     const u16* __restrict__ Wx,   // [3H, Din] bf16 row-major
                     const u16* __restrict__ Wh,   // [3H, H]   bf16 row-major
                     const float* __restrict__ bx, // [3H]
                     const float* __restrict__ bh, // [3H]
                     const u16* __restrict__ hprev,// [B, H] bf16
                     u16* __restrict__ hnew)       // [B, H] bf16
{
    const int wave = threadIdx.x >> 5;
    const int lane = threadIdx.x & 31;
    const int tile = blockIdx.x * 8 + wave;   // 0..255
    const int m0   = (tile >> 6) << 4;        // batch-row base
    const int j0   = (tile & 63) << 4;        // hidden-col base
    const int lm   = lane & 15;               // row/col within tile
    const int kh   = lane >> 4;               // K-half selector
    const int kofs = kh << 4;                 // 0 or 16

    v8f rs = {}, zs = {}, ni = {}, nh = {};

    // Phase 1: x_t @ Wx^T  -> r, z, i_n
    {
        const u16* arow = xsrc + (size_t)(m0 + lm) * xRowStride;
        const u16* wr   = Wx + (size_t)(0 * Hdim + j0 + lm) * Din;
        const u16* wz   = Wx + (size_t)(1 * Hdim + j0 + lm) * Din;
        const u16* wn   = Wx + (size_t)(2 * Hdim + j0 + lm) * Din;
        gemm_phase(arow, wr, wz, wn, Din, kofs, rs, zs, ni);
    }
    // Phase 2: h_prev @ Wh^T -> r, z, h_n
    {
        const u16* arow = hprev + (size_t)(m0 + lm) * Hdim;
        const u16* wr   = Wh + (size_t)(0 * Hdim + j0 + lm) * Hdim;
        const u16* wz   = Wh + (size_t)(1 * Hdim + j0 + lm) * Hdim;
        const u16* wn   = Wh + (size_t)(2 * Hdim + j0 + lm) * Hdim;
        gemm_phase(arow, wr, wz, wn, Hdim, kofs, rs, zs, nh);
    }

    // Epilogue. C/D layout: lane -> column j0+lm (fixed), VGPR v -> row
    // m0 + v + 8*(lane/16). Gate math is per-lane elementwise.
    const int   j   = j0 + lm;
    const float brj = bx[j] + bh[j];
    const float bzj = bx[Hdim + j] + bh[Hdim + j];
    const float bij = bx[2 * Hdim + j];
    const float bhj = bh[2 * Hdim + j];

#pragma unroll
    for (int v = 0; v < 8; ++v) {
        const int row = m0 + v + (kh << 3);
        float r    = sigmoidf_fast(rs[v] + brj);
        float z    = sigmoidf_fast(zs[v] + bzj);
        float hn   = nh[v] + bhj;
        float nval = tanhf(ni[v] + bij + r * hn);
        float hold = bf2f(hprev[(size_t)row * Hdim + j]);
        float hout = nval + z * (hold - nval);
        hnew[(size_t)row * Hdim + j] = f2bf(hout);
    }
}

// ---------------------------------------------------------------------------
// FC head: out[b] = h1_final[b,:] . fcW + fcb   (wave32 shuffle reduction)
// ---------------------------------------------------------------------------
__global__ __launch_bounds__(256)
void fc_kernel(const u16* __restrict__ h, const float* __restrict__ fcW,
               const float* __restrict__ fcb, float* __restrict__ out) {
    __shared__ float wsum[8];
    const int b    = blockIdx.x;
    const int lane = threadIdx.x & 31;
    const int wid  = threadIdx.x >> 5;

    float s = 0.0f;
    for (int j = threadIdx.x; j < Hdim; j += 256)
        s += bf2f(h[(size_t)b * Hdim + j]) * fcW[j];

    // intra-wave reduction (wave32)
    for (int off = 16; off > 0; off >>= 1)
        s += __shfl_xor(s, off, 32);
    if (lane == 0) wsum[wid] = s;
    __syncthreads();
    if (threadIdx.x == 0) {
        float t = 0.0f;
#pragma unroll
        for (int w = 0; w < 8; ++w) t += wsum[w];
        out[b] = t + fcb[0];
    }
}

// ---------------------------------------------------------------------------
// Host launcher
// ---------------------------------------------------------------------------
extern "C" void kernel_launch(void* const* d_in, const int* in_sizes, int n_in,
                              void* d_out, int out_size, void* d_ws, size_t ws_size,
                              hipStream_t stream) {
    (void)in_sizes; (void)n_in; (void)out_size; (void)ws_size;

    const float* x   = (const float*)d_in[0];   // [B,T,D]
    const float* Wx0 = (const float*)d_in[1];   // [3H,D]
    const float* bx0 = (const float*)d_in[2];   // [3H]
    const float* Wh0 = (const float*)d_in[3];   // [3H,H]
    const float* bh0 = (const float*)d_in[4];   // [3H]
    const float* Wx1 = (const float*)d_in[5];   // [3H,H]
    const float* bx1 = (const float*)d_in[6];
    const float* Wh1 = (const float*)d_in[7];
    const float* bh1 = (const float*)d_in[8];
    const float* fcW = (const float*)d_in[9];   // [1,H]
    const float* fcb = (const float*)d_in[10];  // [1]
    float* out = (float*)d_out;                 // [B,1]

    // Workspace carve-out (256B aligned slices)
    char*  base = (char*)d_ws;
    size_t off  = 0;
    auto alloc = [&](size_t bytes) -> char* {
        char* r = base + off;
        off += (bytes + 255) & ~(size_t)255;
        return r;
    };
    const size_t nX = (size_t)Bdim * Tdim * Ddim;   // 33.5M elems
    const size_t nW = (size_t)3 * Hdim * Ddim;      // 3.1M elems each
    const size_t nH = (size_t)Bdim * Hdim;          // 65536 elems

    u16* Xb   = (u16*)alloc(nX * 2);
    u16* Wx0b = (u16*)alloc(nW * 2);
    u16* Wh0b = (u16*)alloc(nW * 2);
    u16* Wx1b = (u16*)alloc(nW * 2);
    u16* Wh1b = (u16*)alloc(nW * 2);
    u16* h0b[2] = { (u16*)alloc(nH * 2), (u16*)alloc(nH * 2) };
    u16* h1b[2] = { (u16*)alloc(nH * 2), (u16*)alloc(nH * 2) };

    // 1) One-time fp32 -> bf16 conversions
    {
        int n = (int)nX;
        f32_to_bf16_kernel<<<(n + 255) / 256, 256, 0, stream>>>(x, Xb, n);
        n = (int)nW;
        f32_to_bf16_kernel<<<(n + 255) / 256, 256, 0, stream>>>(Wx0, Wx0b, n);
        f32_to_bf16_kernel<<<(n + 255) / 256, 256, 0, stream>>>(Wh0, Wh0b, n);
        f32_to_bf16_kernel<<<(n + 255) / 256, 256, 0, stream>>>(Wx1, Wx1b, n);
        f32_to_bf16_kernel<<<(n + 255) / 256, 256, 0, stream>>>(Wh1, Wh1b, n);
    }
    // 2) Zero initial hidden states (ping buffer 0 of each layer)
    {
        int n = (int)nH;
        zero_u16_kernel<<<(n + 255) / 256, 256, 0, stream>>>(h0b[0], n);
        zero_u16_kernel<<<(n + 255) / 256, 256, 0, stream>>>(h1b[0], n);
    }

    // 3) Recurrence: per step, layer0 then layer1 (stream order gives the dep)
    const int nTiles = (Bdim / 16) * (Hdim / 16);   // 256 waves
    const int nBlks  = nTiles / 8;                  // 32 blocks x 8 waves
    for (int t = 0; t < Tdim; ++t) {
        const u16* h0p = h0b[t & 1];
        u16*       h0n = h0b[(t + 1) & 1];
        const u16* h1p = h1b[t & 1];
        u16*       h1n = h1b[(t + 1) & 1];
        // Layer 0: x input = Xb[:, t, :], batch stride T*D
        gru_step_kernel<<<nBlks, 256, 0, stream>>>(
            Xb + (size_t)t * Ddim, (size_t)Tdim * Ddim, Ddim,
            Wx0b, Wh0b, bx0, bh0, h0p, h0n);
        // Layer 1: x input = layer0's fresh h (batch stride H)
        gru_step_kernel<<<nBlks, 256, 0, stream>>>(
            h0n, (size_t)Hdim, Hdim,
            Wx1b, Wh1b, bx1, bh1, h1p, h1n);
    }

    // 4) FC head on the final layer-1 state (T even -> lands in buffer 0)
    fc_kernel<<<Bdim, 256, 0, stream>>>(h1b[0], fcW, fcb, out);
}
